// MambaModel_34265249087557
// MI455X (gfx1250) — matrix operand order repaired and checked by reading, not compiled, not run on hardware
//
#include <hip/hip_runtime.h>
#include <hip/hip_bf16.h>
#include <math.h>
#include <stdint.h>

typedef __attribute__((ext_vector_type(16))) _Float16 v16h;
typedef __attribute__((ext_vector_type(8)))  _Float16 h8v;
typedef __attribute__((ext_vector_type(8)))  float    v8f;

typedef int v4i __attribute__((__vector_size__(16)));
typedef __attribute__((address_space(1))) v4i g_v4i;
typedef __attribute__((address_space(3))) v4i l_v4i;

namespace {
constexpr int BATCH = 4, LSEQ = 2048, D_IN = 128, DM = 768, NL = 2;
constexpr int ED = 2 * DM;            // 1536
constexpr int NST = 16, DTR = 48, KCONV = 4, D_OUT = 128;
constexpr int MROWS = BATCH * LSEQ;   // 8192
constexpr int DBCW  = DTR + 2 * NST;  // 80
constexpr float EPS = 1e-5f;
}

#if __has_builtin(__builtin_amdgcn_global_load_async_to_lds_b128)
#define HAVE_ASYNC_LDS 1
#else
#define HAVE_ASYNC_LDS 0
#endif

__device__ __forceinline__ float sigmoidf_(float x) { return 1.0f / (1.0f + expf(-x)); }

__device__ __forceinline__ v16h cat16(h8v lo, h8v hi) {
  return __builtin_shufflevector(lo, hi, 0, 1, 2, 3, 4, 5, 6, 7,
                                         8, 9, 10, 11, 12, 13, 14, 15);
}

// 16B global -> LDS copy: async (ASYNCcnt-tracked) when available.
__device__ __forceinline__ void copy_b128_to_lds(const _Float16* g, _Float16* l) {
#if HAVE_ASYNC_LDS
  __builtin_amdgcn_global_load_async_to_lds_b128(
      (g_v4i*)(uintptr_t)g, (l_v4i*)(uintptr_t)l, 0, 0);
#else
  *(h8v*)l = *(const h8v*)g;
#endif
}

__device__ __forceinline__ void async_copies_wait() {
#if HAVE_ASYNC_LDS
#if __has_builtin(__builtin_amdgcn_s_wait_asynccnt)
  __builtin_amdgcn_s_wait_asynccnt(0);
#else
  asm volatile("s_wait_asynccnt 0x0" ::: "memory");
#endif
#endif
}

// ---------------------------------------------------------------------------
// LDS-staged GEMM: D[M,N] = act(A*W^T + bias + Cin), A/W f16, f32 accumulate.
// Block = 8 waves -> 128 rows x 64 cols. B chunks (64 cols x 32 K) staged into
// double-buffered LDS via async global->LDS b128 copies (one 16B transfer per
// thread per chunk); all 8 waves read fragments with ds_load_b128.
// Requires: M%128==0, N%64==0, K%32==0.
// ---------------------------------------------------------------------------
__global__ __launch_bounds__(256) void gemm_wmma_f16_lds(
    const _Float16* __restrict__ A, int lda,
    const _Float16* __restrict__ W, int K,
    const float* __restrict__ bias,
    const float* __restrict__ Cin, int ldc,
    float* __restrict__ D, _Float16* __restrict__ D16, int ldd,
    int M, int N, int act)
{
  constexpr int BN = 64, BK = 32;
  constexpr int SSTR = 40;                 // padded row stride (halves)
  __shared__ _Float16 sB[2][BN * SSTR];

  const int nblk = N / BN;
  const int bm = blockIdx.x / nblk;
  const int bn = blockIdx.x % nblk;
  const int tid = threadIdx.x;
  const int w = tid >> 5;
  const int lane = tid & 31;
  const int hsel = lane >> 4;
  const int lr   = lane & 15;
  const int mt = bm * 8 + w;               // this wave's 16-row tile
  const int colbase = bn * BN;

  // staging role: thread stages 16B of W: column tid/4, 8-half segment tid%4
  const int scol = tid >> 2;
  const int sseg = (tid & 3) * 8;
  const _Float16* __restrict__ wsrc = W + (size_t)(colbase + scol) * K + sseg;
  const int soff = scol * SSTR + sseg;

  const _Float16* __restrict__ arow = A + (size_t)(mt * 16 + lr) * lda;

  v8f acc[4];
  if (Cin) {
#pragma unroll
    for (int i = 0; i < 4; ++i)
#pragma unroll
      for (int r = 0; r < 8; ++r)
        acc[i][r] = Cin[(size_t)(mt * 16 + hsel * 8 + r) * ldc + colbase + i * 16 + lr];
  } else {
    const v8f zf = {};
#pragma unroll
    for (int i = 0; i < 4; ++i) acc[i] = zf;
  }

  // prologue: stage chunk 0
  copy_b128_to_lds(wsrc, &sB[0][soff]);
  async_copies_wait();
  __syncthreads();

  const int nk = K / BK;
  for (int ki = 0; ki < nk; ++ki) {
    const int kk = ki * BK;
    const int buf = ki & 1;
    if (ki + 1 < nk)  // overlap next chunk's copy with this chunk's WMMAs
      copy_b128_to_lds(wsrc + kk + BK, &sB[buf ^ 1][soff]);

    __builtin_prefetch(arow + kk + 128, 0, 3);
    // A 16x32 fragment: lanes 0-15 -> K kk..kk+15; lanes 16-31 -> K kk+16..31
    h8v a0 = *(const h8v*)(arow + kk + hsel * 16);
    h8v a1 = *(const h8v*)(arow + kk + hsel * 16 + 8);
    v16h a = cat16(a0, a1);

    const _Float16* sbuf = sB[buf];
#pragma unroll
    for (int i = 0; i < 4; ++i) {
      // B 32x16 fragment from LDS: lane = column; lanes 0-15 K kk..+15, 16-31 K kk+16..31
      const _Float16* srow = sbuf + (i * 16 + lr) * SSTR + hsel * 16;
      h8v b0 = *(const h8v*)(srow);
      h8v b1 = *(const h8v*)(srow + 8);
      acc[i] = __builtin_amdgcn_wmma_f32_16x16x32_f16(false, a, false, cat16(b0, b1),
                                                      (short)0, acc[i], false, false);
    }
    async_copies_wait();
    __syncthreads();
  }

#pragma unroll
  for (int i = 0; i < 4; ++i) {
    const int col = colbase + i * 16 + lr;
    const float bv = bias ? bias[col] : 0.0f;
#pragma unroll
    for (int r = 0; r < 8; ++r) {
      float v = acc[i][r] + bv;
      if (act == 1) v = (v > 20.0f) ? v : log1pf(expf(v));
      const size_t idx = (size_t)(mt * 16 + hsel * 8 + r) * ldd + col;
      if (D)   D[idx]   = v;
      if (D16) D16[idx] = (_Float16)v;
    }
  }
}

static void launch_gemm_lds(const _Float16* A, int lda, const _Float16* W, int K,
                            const float* bias, const float* Cin, int ldc,
                            float* D, _Float16* D16, int ldd, int Mr, int N, int act,
                            hipStream_t stream)
{
  const int blocks = (Mr / 128) * (N / 64);
  gemm_wmma_f16_lds<<<blocks, 256, 0, stream>>>(A, lda, W, K, bias, Cin, ldc,
                                                D, D16, ldd, Mr, N, act);
}

// ---------------------------------------------------------------------------
// Register-path GEMM for irregular shapes (N=80 -> NTW=5; K=48 tail -> NTW=4).
// ---------------------------------------------------------------------------
template <int NTW>
__global__ __launch_bounds__(256) void gemm_wmma_f16_reg(
    const _Float16* __restrict__ A, int lda,
    const _Float16* __restrict__ W, int K,
    const float* __restrict__ bias,
    float* __restrict__ D, _Float16* __restrict__ D16, int ldd,
    int M, int N, int act)
{
  const int ngrp = N / (16 * NTW);
  const int mtiles = M >> 4;
  const int wave = blockIdx.x * (blockDim.x >> 5) + (threadIdx.x >> 5);
  if (wave >= mtiles * ngrp) return;
  const int mt = wave / ngrp;
  const int ng = wave % ngrp;
  const int lane = threadIdx.x & 31;
  const int hsel = lane >> 4;
  const int lr   = lane & 15;
  const int colbase = ng * 16 * NTW;

  const _Float16* __restrict__ arow = A + (size_t)(mt * 16 + lr) * lda;

  v8f acc[NTW];
  const v8f zf = {};
#pragma unroll
  for (int i = 0; i < NTW; ++i) acc[i] = zf;

  const int Kfull = K & ~31;
  for (int kk = 0; kk < Kfull; kk += 32) {
    __builtin_prefetch(arow + kk + 128, 0, 3);
    h8v a0 = *(const h8v*)(arow + kk + hsel * 16);
    h8v a1 = *(const h8v*)(arow + kk + hsel * 16 + 8);
    v16h a = cat16(a0, a1);
#pragma unroll
    for (int i = 0; i < NTW; ++i) {
      const _Float16* wrow = W + (size_t)(colbase + i * 16 + lr) * K;
      h8v b0 = *(const h8v*)(wrow + kk + hsel * 16);
      h8v b1 = *(const h8v*)(wrow + kk + hsel * 16 + 8);
      acc[i] = __builtin_amdgcn_wmma_f32_16x16x32_f16(false, a, false, cat16(b0, b1),
                                                      (short)0, acc[i], false, false);
    }
  }
  if (K & 31) {  // K%32==16 tail (K=48 dt GEMM): zero-pad upper half of chunk
    const int kk = Kfull;
    const h8v z = {};
    h8v a0 = *(const h8v*)(arow + kk);  // kk..kk+15, all valid (B zero for lanes 16-31)
    v16h a = cat16(a0, *(const h8v*)(arow + kk + 8));
#pragma unroll
    for (int i = 0; i < NTW; ++i) {
      h8v b0 = z, b1 = z;
      if (hsel == 0) {  // lanes 16-31 would need K kk+16..31 -> zero
        const _Float16* wrow = W + (size_t)(colbase + i * 16 + lr) * K;
        b0 = *(const h8v*)(wrow + kk);
        b1 = *(const h8v*)(wrow + kk + 8);
      }
      acc[i] = __builtin_amdgcn_wmma_f32_16x16x32_f16(false, a, false, cat16(b0, b1),
                                                      (short)0, acc[i], false, false);
    }
  }

#pragma unroll
  for (int i = 0; i < NTW; ++i) {
    const int col = colbase + i * 16 + lr;
    const float bv = bias ? bias[col] : 0.0f;
#pragma unroll
    for (int r = 0; r < 8; ++r) {
      float v = acc[i][r] + bv;
      if (act == 1) v = (v > 20.0f) ? v : log1pf(expf(v));
      const size_t idx = (size_t)(mt * 16 + hsel * 8 + r) * ldd + col;
      if (D)   D[idx]   = v;
      if (D16) D16[idx] = (_Float16)v;
    }
  }
}

template <int NTW>
static void launch_gemm_reg(const _Float16* A, int lda, const _Float16* W, int K,
                            const float* bias, float* D, _Float16* D16, int ldd,
                            int Mr, int N, int act, hipStream_t stream)
{
  const long waves = (long)(Mr / 16) * (N / (16 * NTW));
  const int blocks = (int)((waves + 7) / 8);
  gemm_wmma_f16_reg<NTW><<<blocks, 256, 0, stream>>>(A, lda, W, K, bias,
                                                     D, D16, ldd, Mr, N, act);
}

__global__ __launch_bounds__(256) void f32_to_f16_kernel(
    const float* __restrict__ src, _Float16* __restrict__ dst, long n)
{
  const long g = (long)blockIdx.x * blockDim.x + threadIdx.x;
  if (g < n) dst[g] = (_Float16)src[g];
}

// xn16[row,:] = f16( h[row,:] * rsqrt(mean(h^2)+eps) * w[:] )
__global__ __launch_bounds__(256) void rmsnorm_kernel(
    const float* __restrict__ h, const float* __restrict__ w, _Float16* __restrict__ xn)
{
  const int row = blockIdx.x;
  const float* hr = h + (size_t)row * DM;
  float s = 0.f;
  for (int i = threadIdx.x; i < DM; i += 256) { float v = hr[i]; s += v * v; }
#pragma unroll
  for (int off = 16; off > 0; off >>= 1) s += __shfl_xor(s, off, 32);
  __shared__ float red[8];
  __shared__ float rs_s;
  if ((threadIdx.x & 31) == 0) red[threadIdx.x >> 5] = s;
  __syncthreads();
  if (threadIdx.x == 0) {
    float t = 0.f;
#pragma unroll
    for (int i = 0; i < 8; ++i) t += red[i];
    rs_s = rsqrtf(t / (float)DM + EPS);
  }
  __syncthreads();
  const float rs = rs_s;
  for (int i = threadIdx.x; i < DM; i += 256)
    xn[(size_t)row * DM + i] = (_Float16)(hr[i] * rs * w[i]);
}

// xc16[b,t,e] = f16( silu( sum_j xa[b, t-3+j, e]*cw[e,j] + cb[e] ) ),  xa = xz[..., :ED]
__global__ __launch_bounds__(256) void conv_silu_kernel(
    const float* __restrict__ xz, const float* __restrict__ cw,
    const float* __restrict__ cb, _Float16* __restrict__ xc)
{
  const long gid = (long)blockIdx.x * blockDim.x + threadIdx.x;
  if (gid >= (long)MROWS * ED) return;
  const int  e   = (int)(gid % ED);
  const long row = gid / ED;          // b*L + t
  const int  t   = (int)(row % LSEQ);
  float acc = cb[e];
#pragma unroll
  for (int j = 0; j < KCONV; ++j) {
    const int tt = t - (KCONV - 1) + j;
    if (tt >= 0)
      acc += xz[(size_t)(row + (tt - t)) * (2 * ED) + e] * cw[e * KCONV + j];
  }
  xc[gid] = (_Float16)(acc * sigmoidf_(acc));
}

// Sequential selective scan, one thread per (b,e) channel; B/C broadcast via LDS.
__global__ __launch_bounds__(256) void scan_kernel(
    const _Float16* __restrict__ xc, const _Float16* __restrict__ delta,
    const _Float16* __restrict__ dbc, const float* __restrict__ xz,
    const float* __restrict__ A_log, const float* __restrict__ Dp,
    _Float16* __restrict__ out)
{
  const int b = blockIdx.y;
  const int e = blockIdx.x * 256 + threadIdx.x;
  __shared__ float sBC[32];
  float Av[NST];
#pragma unroll
  for (int n = 0; n < NST; ++n) Av[n] = -expf(A_log[(size_t)e * NST + n]);
  const float dp = Dp[e];
  float hst[NST];
#pragma unroll
  for (int n = 0; n < NST; ++n) hst[n] = 0.f;
  const size_t rb = (size_t)b * LSEQ;
  for (int t = 0; t < LSEQ; ++t) {
    const size_t r = rb + t;
    if (threadIdx.x < 32) sBC[threadIdx.x] = (float)dbc[r * DBCW + DTR + threadIdx.x];
    __syncthreads();
    const float d  = (float)delta[r * ED + e];
    const float xv = (float)xc[r * ED + e];
    const float zv = xz[r * (2 * ED) + ED + e];
    const float dx = d * xv;
    float y = 0.f;
#pragma unroll
    for (int n = 0; n < NST; ++n) {
      const float dA = expf(d * Av[n]);
      hst[n] = dA * hst[n] + dx * sBC[n];
      y += hst[n] * sBC[NST + n];
    }
    y += dp * xv;
    out[r * ED + e] = (_Float16)(y * (zv * sigmoidf_(zv)));
    __syncthreads();
  }
}

extern "C" void kernel_launch(void* const* d_in, const int* in_sizes, int n_in,
                              void* d_out, int out_size, void* d_ws, size_t ws_size,
                              hipStream_t stream)
{
  const float* x       = (const float*)d_in[0];
  const float* lin_w   = (const float*)d_in[1];
  const float* lin_b   = (const float*)d_in[2];
  const float* norm_w  = (const float*)d_in[3];
  const float* in_proj = (const float*)d_in[4];
  const float* conv_w  = (const float*)d_in[5];
  const float* conv_b  = (const float*)d_in[6];
  const float* x_proj  = (const float*)d_in[7];
  const float* dt_w    = (const float*)d_in[8];
  const float* dt_b    = (const float*)d_in[9];
  const float* A_log   = (const float*)d_in[10];
  const float* Dp      = (const float*)d_in[11];
  const float* out_w   = (const float*)d_in[12];
  const float* fc_w    = (const float*)d_in[13];
  const float* fc_b    = (const float*)d_in[14];
  float* outp = (float*)d_out;
  (void)in_sizes; (void)n_in; (void)out_size; (void)ws_size;

  const size_t M = MROWS;
  char* p = (char*)d_ws;
  auto alloc_f32 = [&](size_t n) { float* q = (float*)p; p += n * 4; return q; };
  auto alloc_f16 = [&](size_t n) { _Float16* q = (_Float16*)p; p += n * 2; return q; };

  // fp32 activations
  float* h_buf  = alloc_f32(M * DM);          // residual stream
  float* xz_buf = alloc_f32(M * 2 * ED);      // in-proj output (xa | z)
  // fp16 activations (GEMM A operands)
  _Float16* x16     = alloc_f16(M * D_IN);
  _Float16* xn16    = alloc_f16(M * DM);
  _Float16* xc16    = alloc_f16(M * ED);
  _Float16* dbc16   = alloc_f16(M * DBCW);
  _Float16* delta16 = alloc_f16(M * ED);      // also scan output (in-place)
  _Float16* h16     = alloc_f16(M * DM);
  // fp16 weights
  _Float16* lin_w16  = alloc_f16((size_t)DM * D_IN);
  _Float16* inpj_w16 = alloc_f16((size_t)NL * 2 * ED * DM);
  _Float16* xpj_w16  = alloc_f16((size_t)NL * DBCW * ED);
  _Float16* dt_w16   = alloc_f16((size_t)NL * ED * DTR);
  _Float16* out_w16  = alloc_f16((size_t)NL * DM * ED);
  _Float16* fc_w16   = alloc_f16((size_t)D_OUT * DM);

  auto cvt = [&](const float* src, _Float16* dst, long n) {
    f32_to_f16_kernel<<<(int)((n + 255) / 256), 256, 0, stream>>>(src, dst, n);
  };
  cvt(x,       x16,      (long)M * D_IN);
  cvt(lin_w,   lin_w16,  (long)DM * D_IN);
  cvt(in_proj, inpj_w16, (long)NL * 2 * ED * DM);
  cvt(x_proj,  xpj_w16,  (long)NL * DBCW * ED);
  cvt(dt_w,    dt_w16,   (long)NL * ED * DTR);
  cvt(out_w,   out_w16,  (long)NL * DM * ED);
  cvt(fc_w,    fc_w16,   (long)D_OUT * DM);

  // h = x @ lin_w^T + lin_b
  launch_gemm_lds(x16, D_IN, lin_w16, D_IN, lin_b, nullptr, 0,
                  h_buf, nullptr, DM, MROWS, DM, 0, stream);

  for (int l = 0; l < NL; ++l) {
    rmsnorm_kernel<<<MROWS, 256, 0, stream>>>(h_buf, norm_w + (size_t)l * DM, xn16);
    // xz = xn @ in_proj^T
    launch_gemm_lds(xn16, DM, inpj_w16 + (size_t)l * 2 * ED * DM, DM,
                    nullptr, nullptr, 0, xz_buf, nullptr, 2 * ED, MROWS, 2 * ED, 0, stream);
    // xc = silu(causal_conv(xa))
    const long nconv = (long)MROWS * ED;
    conv_silu_kernel<<<(int)((nconv + 255) / 256), 256, 0, stream>>>(
        xz_buf, conv_w + (size_t)l * ED * KCONV, conv_b + (size_t)l * ED, xc16);
    // dbc = xc @ x_proj^T   (N=80 -> register path, 5 tiles/wave)
    launch_gemm_reg<5>(xc16, ED, xpj_w16 + (size_t)l * DBCW * ED, ED,
                       nullptr, nullptr, dbc16, DBCW, MROWS, DBCW, 0, stream);
    // delta = softplus(dbc[:, :48] @ dt_w^T + dt_b)   (K=48 -> register path w/ tail)
    launch_gemm_reg<4>(dbc16, DBCW, dt_w16 + (size_t)l * ED * DTR, DTR,
                       dt_b + (size_t)l * ED, nullptr, delta16, ED, MROWS, ED, 1, stream);
    // selective scan + gating by silu(z)
    dim3 sg(ED / 256, BATCH);
    scan_kernel<<<sg, 256, 0, stream>>>(xc16, delta16, dbc16, xz_buf,
                                        A_log + (size_t)l * ED * NST,
                                        Dp + (size_t)l * ED, delta16);
    // h += scan @ out_w^T  (residual accumulate through WMMA C operand; also emit h16)
    launch_gemm_lds(delta16, ED, out_w16 + (size_t)l * DM * ED, ED,
                    nullptr, h_buf, DM, h_buf, h16, DM, MROWS, DM, 0, stream);
  }

  // out = h @ fc_w^T + fc_b
  launch_gemm_lds(h16, DM, fc_w16, DM, fc_b, nullptr, 0,
                  outp, nullptr, D_OUT, MROWS, D_OUT, 0, stream);
}
